// MultiHeadSelfAttention_32134945309307
// MI455X (gfx1250) — matrix operand order repaired and checked by reading, not compiled, not run on hardware
//
#include <hip/hip_runtime.h>

// ---------------------------------------------------------------------------
// MultiHeadSelfAttention for MI455X (gfx1250).
//   B=2, S=4096, D_MODEL=512, H=8, d_k=64  ->  ~86 GFLOP, ~45 MB I/O.
// Compute-bound => f16 WMMA (f32 accumulate). Flash-attention streaming so the
// 2.1 GB logits tensor never exists. All operands pre-converted to f16 with
// WMMA-friendly layouts so every fragment load is two global_load_b128 /
// ds_load_b128 (no per-element cvt in the hot loops).
// ---------------------------------------------------------------------------

#define DM    512
#define HEADS 8
#define DK    64
#define SEQ   4096
#define BATCH 2

typedef __attribute__((ext_vector_type(16))) _Float16 v16h;
typedef __attribute__((ext_vector_type(8)))  _Float16 v8h;
typedef __attribute__((ext_vector_type(8)))  float    v8f;

static __device__ inline v8f wmma_f16(v16h a, v16h b, v8f c) {
  // D(16x16,f32) = A(16x32,f16) * B(32x16,f16) + C
  return __builtin_amdgcn_wmma_f32_16x16x32_f16(
      /*neg_a=*/false, a, /*neg_b=*/false, b,
      /*c_mod=*/(short)0, c, /*reuse_a=*/false, /*reuse_b=*/false);
}

static __device__ inline v8f vzero8() {
  v8f z;
#pragma unroll
  for (int i = 0; i < 8; ++i) z[i] = 0.0f;
  return z;
}

static __device__ inline v16h combine16(v8h lo, v8h hi) {
  v16h r;
#pragma unroll
  for (int e = 0; e < 8; ++e) { r[e] = lo[e]; r[e + 8] = hi[e]; }
  return r;
}

// A-fragment (16x32 MxK, f16) from f16 row-major memory.
// ISA 7.12.2: lane holds row M=lane&15; lane-group g=lane>>4:
//   e<8  -> K = e + 8g      (8 contiguous halves at +8g)
//   e>=8 -> K = e + 8 + 8g  (8 contiguous halves at +16+8g)
// rowp = &A[row][kk]; both runs are 16B aligned.
static __device__ inline v16h load_a_from_f16(const _Float16* rowp, int g) {
  v8h lo = *(const v8h*)(rowp + 8 * g);
  v8h hi = *(const v8h*)(rowp + 16 + 8 * g);
  return combine16(lo, hi);
}

// B-fragment (32x16 KxN, f16) where the contraction index is contiguous in
// memory (matrix stored "col-major"/transposed): lane holds column N=lane&15,
// element e<8 -> K = e + 8g, e>=8 -> K = e + 8 + 8g.  colp = &Bt[col][kk]. Two b128 loads.
static __device__ inline v16h load_b_f16_contig(const _Float16* colp, int g) {
  v8h lo = *(const v8h*)(colp + 8 * g);
  v8h hi = *(const v8h*)(colp + 16 + 8 * g);
  return combine16(lo, hi);
}

// ---------------------------------------------------------------------------
// Kernel 0: one-shot conversion (21 MB, bandwidth-trivial).
//   xh[i]          = f16(x[i])                        row-major [8192][512]
//   wt[w][n][k]    = f16(W_w[k][n])   (transposed)    w in {q,k,v,o}
// ---------------------------------------------------------------------------
__global__ void __launch_bounds__(256)
k_convert(const float* __restrict__ x,  const float* __restrict__ Wq,
          const float* __restrict__ Wk, const float* __restrict__ Wv,
          const float* __restrict__ Wo,
          _Float16* __restrict__ xh, _Float16* __restrict__ wt) {
  size_t i = (size_t)blockIdx.x * blockDim.x + threadIdx.x;
  const size_t N1 = (size_t)BATCH * SEQ * DM;  // 4,194,304
  if (i < N1) {
    xh[i] = (_Float16)x[i];
    return;
  }
  size_t j = i - N1;                 // 0 .. 4*DM*DM-1
  int w   = (int)(j >> 18);          // DM*DM = 2^18
  int rem = (int)(j & (DM * DM - 1));
  int n = rem >> 9, k = rem & (DM - 1);
  const float* W = (w == 0) ? Wq : (w == 1) ? Wk : (w == 2) ? Wv : Wo;
  wt[j] = (_Float16)W[(size_t)k * DM + n];
}

// ---------------------------------------------------------------------------
// Kernel 1: QKV projection.  grid=(M/16, DM/64, 3), block=32 (one wave).
// Wave computes a 16x64 strip of xh @ W{q|k|v}. Inner loop per 32-K step:
// 2 b128 (A) + 8 b128 (B) + 4 WMMA. Writes f16:
//   z=0 -> Q[b][h][s][d], z=1 -> K[b][h][s][d], z=2 -> V^T[b][h][d][s]
// (V transposed so the PV B-fragment is contiguous in the key dimension.)
// ---------------------------------------------------------------------------
__global__ void __launch_bounds__(32)
k_proj_qkv(const _Float16* __restrict__ xh, const _Float16* __restrict__ wt,
           _Float16* __restrict__ q_ws, _Float16* __restrict__ k_ws,
           _Float16* __restrict__ vt_ws) {
  const int lane = threadIdx.x & 31;
  const int g = lane >> 4, c16 = lane & 15;
  const int m0 = blockIdx.x * 16;
  const int n0 = blockIdx.y * 64;
  const int z  = blockIdx.z;
  const _Float16* Wz = wt + (size_t)z * DM * DM;  // [n][k] f16

  v8f acc[4] = {vzero8(), vzero8(), vzero8(), vzero8()};

  for (int kk = 0; kk < DM; kk += 32) {
    v16h a = load_a_from_f16(xh + (size_t)(m0 + c16) * DM + kk, g);
#pragma unroll
    for (int t = 0; t < 4; ++t) {
      v16h b = load_b_f16_contig(Wz + (size_t)(n0 + 16 * t + c16) * DM + kk, g);
      acc[t] = wmma_f16(a, b, acc[t]);
    }
  }

  // D layout: VGPR r -> row m0 + r + 8g; lane c16 -> col n0 + 16t + c16
#pragma unroll
  for (int t = 0; t < 4; ++t) {
#pragma unroll
    for (int r = 0; r < 8; ++r) {
      int mm = m0 + r + 8 * g;
      int nn = n0 + 16 * t + c16;
      int bb = mm >> 12, ss = mm & (SEQ - 1);
      int hh = nn >> 6,  dd = nn & (DK - 1);
      _Float16 val = (_Float16)acc[t][r];
      if (z == 0)
        q_ws[((size_t)(bb * HEADS + hh) * SEQ + ss) * DK + dd] = val;
      else if (z == 1)
        k_ws[((size_t)(bb * HEADS + hh) * SEQ + ss) * DK + dd] = val;
      else
        vt_ws[((size_t)(bb * HEADS + hh) * DK + dd) * SEQ + ss] = val;
    }
  }
}

// ---------------------------------------------------------------------------
// Kernel 2: flash attention.  grid=(SEQ/32, HEADS, BATCH), block=32.
// One wave owns 32 queries (two 16-row A tiles) of one (b,h): each K/V
// fragment load is amortized over two Q tiles (2x FLOP/byte vs 16-query).
// Per 32-key chunk: 16 b128 global loads, 16 WMMAs, online softmax in f32.
// P tiles round-trip through LDS to convert C-layout -> A-layout.
// Output: concat O[b][s][h*64+d] as f16 (feeds the final f16 GEMM).
// ---------------------------------------------------------------------------
__global__ void __launch_bounds__(32)
k_attn(const _Float16* __restrict__ q_ws, const _Float16* __restrict__ k_ws,
       const _Float16* __restrict__ vt_ws, _Float16* __restrict__ oh_ws) {
  __shared__ __align__(16) _Float16 pbuf[2][16 * 32];  // per Q-tile: 16q x 32k

  const int lane = threadIdx.x & 31;
  const int g = lane >> 4, c16 = lane & 15;
  const int q0 = blockIdx.x * 32;
  const int h = blockIdx.y, b = blockIdx.z;
  const int bh = b * HEADS + h;
  const _Float16* Q  = q_ws  + (size_t)bh * SEQ * DK;
  const _Float16* Km = k_ws  + (size_t)bh * SEQ * DK;
  const _Float16* Vt = vt_ws + (size_t)bh * DK * SEQ;

  // Resident Q fragments: qa[u][kk] = queries q0+16u.., d-range kk*32..
  v16h qa[2][2];
#pragma unroll
  for (int u = 0; u < 2; ++u)
#pragma unroll
    for (int kk = 0; kk < 2; ++kk)
      qa[u][kk] = load_a_from_f16(Q + (size_t)(q0 + 16 * u + c16) * DK + kk * 32, g);

  float mrow[2][8], lrow[2][8];
  v8f oacc[2][4];
#pragma unroll
  for (int u = 0; u < 2; ++u) {
#pragma unroll
    for (int r = 0; r < 8; ++r) { mrow[u][r] = -3.0e38f; lrow[u][r] = 0.0f; }
#pragma unroll
    for (int t = 0; t < 4; ++t) oacc[u][t] = vzero8();
  }
  const float scale = 0.125f;  // 1/sqrt(d_k)

  for (int c = 0; c < SEQ; c += 32) {
    // --- S = Q K^T : 32 keys (two 16-key N tiles) x two Q tiles ---
    v8f s[2][2] = {{vzero8(), vzero8()}, {vzero8(), vzero8()}};
#pragma unroll
    for (int kk = 0; kk < 2; ++kk) {
      v16h b0 = load_b_f16_contig(Km + (size_t)(c + c16) * DK + kk * 32, g);
      v16h b1 = load_b_f16_contig(Km + (size_t)(c + 16 + c16) * DK + kk * 32, g);
#pragma unroll
      for (int u = 0; u < 2; ++u) {
        s[u][0] = wmma_f16(qa[u][kk], b0, s[u][0]);
        s[u][1] = wmma_f16(qa[u][kk], b1, s[u][1]);
      }
    }

    // --- online softmax per Q tile; row M = r + 8g lives in one 16-lane half
    float fac[2][8];
#pragma unroll
    for (int u = 0; u < 2; ++u) {
#pragma unroll
      for (int r = 0; r < 8; ++r) {
        float l0 = s[u][0][r] * scale, l1 = s[u][1][r] * scale;
        float mx = fmaxf(l0, l1);
        mx = fmaxf(mx, __shfl_xor(mx, 1, 32));
        mx = fmaxf(mx, __shfl_xor(mx, 2, 32));
        mx = fmaxf(mx, __shfl_xor(mx, 4, 32));
        mx = fmaxf(mx, __shfl_xor(mx, 8, 32));
        float nm = fmaxf(mrow[u][r], mx);
        fac[u][r] = __expf(mrow[u][r] - nm);
        float p0 = __expf(l0 - nm);
        float p1 = __expf(l1 - nm);
        float rs = p0 + p1;
        rs += __shfl_xor(rs, 1, 32);
        rs += __shfl_xor(rs, 2, 32);
        rs += __shfl_xor(rs, 4, 32);
        rs += __shfl_xor(rs, 8, 32);
        lrow[u][r] = lrow[u][r] * fac[u][r] + rs;
        mrow[u][r] = nm;
        pbuf[u][(r + 8 * g) * 32 + c16]      = (_Float16)p0;
        pbuf[u][(r + 8 * g) * 32 + 16 + c16] = (_Float16)p1;
      }
      // rescale running O by exp(m_old - m_new)
#pragma unroll
      for (int t = 0; t < 4; ++t)
#pragma unroll
        for (int r = 0; r < 8; ++r) oacc[u][t][r] *= fac[u][r];
    }

    __syncthreads();  // cross-half LDS exchange (single-wave WG: waits + NOP)

    // reload P tiles as A-fragments (two ds_load_b128 each)
    v16h pa[2];
#pragma unroll
    for (int u = 0; u < 2; ++u) {
      const _Float16* rowp = &pbuf[u][c16 * 32];
      v8h lo = *(const v8h*)(rowp + 8 * g);
      v8h hi = *(const v8h*)(rowp + 16 + 8 * g);
      pa[u] = combine16(lo, hi);
    }

    // --- O += P V : contraction over 32 keys; V fragment shared by both u ---
#pragma unroll
    for (int t = 0; t < 4; ++t) {
      v16h vb = load_b_f16_contig(Vt + (size_t)(16 * t + c16) * SEQ + c, g);
#pragma unroll
      for (int u = 0; u < 2; ++u) oacc[u][t] = wmma_f16(pa[u], vb, oacc[u][t]);
    }
    __syncthreads();  // WAR: next chunk's P stores behind this chunk's loads
  }

  // epilogue: O /= l ; write concat layout [b][s][h*64+d] as f16
#pragma unroll
  for (int u = 0; u < 2; ++u) {
    float inv[8];
#pragma unroll
    for (int r = 0; r < 8; ++r) inv[r] = 1.0f / lrow[u][r];
#pragma unroll
    for (int t = 0; t < 4; ++t) {
#pragma unroll
      for (int r = 0; r < 8; ++r) {
        int ss = q0 + 16 * u + r + 8 * g;
        oh_ws[(size_t)(b * SEQ + ss) * DM + h * DK + 16 * t + c16] =
            (_Float16)(oacc[u][t][r] * inv[r]);
      }
    }
  }
}

// ---------------------------------------------------------------------------
// Kernel 3: output projection  out = O_concat(f16) @ W_o  -> f32
// grid=(M/16, DM/64), block=32. Same fragment scheme as kernel 1.
// ---------------------------------------------------------------------------
__global__ void __launch_bounds__(32)
k_gemm_out(const _Float16* __restrict__ oh, const _Float16* __restrict__ wo_t,
           float* __restrict__ out) {
  const int lane = threadIdx.x & 31;
  const int g = lane >> 4, c16 = lane & 15;
  const int m0 = blockIdx.x * 16;
  const int n0 = blockIdx.y * 64;

  v8f acc[4] = {vzero8(), vzero8(), vzero8(), vzero8()};

  for (int kk = 0; kk < DM; kk += 32) {
    v16h a = load_a_from_f16(oh + (size_t)(m0 + c16) * DM + kk, g);
#pragma unroll
    for (int t = 0; t < 4; ++t) {
      v16h b = load_b_f16_contig(wo_t + (size_t)(n0 + 16 * t + c16) * DM + kk, g);
      acc[t] = wmma_f16(a, b, acc[t]);
    }
  }

#pragma unroll
  for (int t = 0; t < 4; ++t) {
#pragma unroll
    for (int r = 0; r < 8; ++r) {
      int mm = m0 + r + 8 * g;
      int nn = n0 + 16 * t + c16;
      out[(size_t)mm * DM + nn] = acc[t][r];
    }
  }
}

// ---------------------------------------------------------------------------
extern "C" void kernel_launch(void* const* d_in, const int* in_sizes, int n_in,
                              void* d_out, int out_size, void* d_ws, size_t ws_size,
                              hipStream_t stream) {
  (void)in_sizes; (void)n_in; (void)out_size; (void)ws_size;
  const float* x  = (const float*)d_in[0];
  const float* Wq = (const float*)d_in[1];
  const float* Wk = (const float*)d_in[2];
  const float* Wv = (const float*)d_in[3];
  const float* Wo = (const float*)d_in[4];
  float* out = (float*)d_out;

  // Workspace partition (f16 halves; ~42 MiB total):
  //   xh  [8192][512]          4,194,304
  //   wt  4 x [512][512]^T     1,048,576
  //   Q   [B][H][S][64]        4,194,304
  //   K   [B][H][S][64]        4,194,304
  //   V^T [B][H][64][S]        4,194,304
  //   Oh  [8192][512]          4,194,304
  const size_t NX = (size_t)BATCH * SEQ * DM;       // 4,194,304
  const size_t NW = (size_t)4 * DM * DM;            // 1,048,576
  _Float16* xh    = (_Float16*)d_ws;
  _Float16* wt    = xh + NX;
  _Float16* q_ws  = wt + NW;
  _Float16* k_ws  = q_ws + NX;
  _Float16* vt_ws = k_ws + NX;
  _Float16* oh_ws = vt_ws + NX;

  const size_t NCONV = NX + NW;                     // 5,242,880 (mult of 256)
  k_convert<<<dim3((unsigned)(NCONV / 256)), 256, 0, stream>>>(
      x, Wq, Wk, Wv, Wo, xh, wt);

  dim3 g1(BATCH * SEQ / 16, DM / 64, 3);
  k_proj_qkv<<<g1, 32, 0, stream>>>(xh, wt, q_ws, k_ws, vt_ws);

  dim3 g2(SEQ / 32, HEADS, BATCH);
  k_attn<<<g2, 32, 0, stream>>>(q_ws, k_ws, vt_ws, oh_ws);

  dim3 g3(BATCH * SEQ / 16, DM / 64);
  k_gemm_out<<<g3, 32, 0, stream>>>(oh_ws, wt + (size_t)3 * DM * DM, out);
}
